// SoftKNN_41154376630931
// MI455X (gfx1250) — matrix operand order repaired and checked by reading, not compiled, not run on hardware
//
#include <hip/hip_runtime.h>

typedef __attribute__((ext_vector_type(16))) __bf16 v16bf;
typedef __attribute__((ext_vector_type(8)))  __bf16 v8bf;
typedef __attribute__((ext_vector_type(8)))  float  v8f;
typedef __attribute__((address_space(3))) void lds_void_t;

#define K_CL   65536
#define D_DIM  256
#define O_DIM  256
#define B_ROWS 1024
#define TOPK   32

#define WAVES_PER_BLK  4
#define ROWS_PER_BLK   64                  // 4 waves x 16-row m-tiles
#define KSB            32                  // cluster splits (gridDim.y)
#define STAGES         (K_CL / KSB / 16)   // 128 16-cluster stages per block
#define NLISTS         (KSB * 2)           // 64 candidate lists per batch row

// workspace layout (bytes)
//   Apack: per cluster row k (2KB): [ivh(512B)][ivl][mivh][mivl], bf16
#define OFF_A     ((size_t)0)              // 65536 * 2048 = 128 MB
#define OFF_CC    ((size_t)134217728)      // 256 KB
#define OFF_CAND  ((size_t)134479872)      // 16 MB

// ---------------------------------------------------------------------------
__device__ inline v8f wmma_bf16(v16bf a, v16bf b, v8f c) {
  return __builtin_amdgcn_wmma_f32_16x16x32_bf16(false, a, false, b, (short)0, c,
                                                 false, false);
}

// A-fragment (16-bit A 16x32 layout): two contiguous 8-elem runs per lane.
__device__ inline v16bf loadA(const __bf16* p) {
  v8bf a = *(const v8bf*)p;
  v8bf b = *(const v8bf*)(p + 16);
  return __builtin_shufflevector(a, b, 0,1,2,3,4,5,6,7,8,9,10,11,12,13,14,15);
}

__device__ inline unsigned lds_byte_off(const void* p) {
  return (unsigned)(unsigned long long)(const lds_void_t*)p;
}

// One stage = 32KB = 2048 x 16B chunks; 128 threads x 16 async b128 each.
// Same IOFFSET applies to both LDS and global sides -> identical layouts.
__device__ inline void issue_async_stage(unsigned lds_off, unsigned voff,
                                         unsigned long long gbase) {
  #pragma unroll
  for (int i = 0; i < 16; ++i) {
    asm volatile("global_load_async_to_lds_b128 %0, %1, %2 offset:%3"
                 :: "v"(lds_off), "v"(voff), "s"(gbase), "i"(i * 2048)
                 : "memory");
  }
}

__device__ inline void topk_ins(uint2* slots, float v, int idx,
                                float& rmin, int& rslot) {
  if (v > rmin) {
    slots[rslot] = make_uint2(__float_as_uint(v), (unsigned)idx);
    float mn = __uint_as_float(slots[0].x);
    int ms = 0;
    #pragma unroll
    for (int s = 1; s < TOPK; ++s) {
      float sv = __uint_as_float(slots[s].x);
      if (sv < mn) { mn = sv; ms = s; }
    }
    rmin = mn; rslot = ms;
  }
}

// ---------------------------------------------------------------------------
// Kernel 1: pack bf16 hi/lo of inv_var and mean*inv_var, interleaved per
// cluster row (2KB/row, TDM-friendly), plus c_k = -0.5*sum(m^2 iv) - sum(log s)
// ---------------------------------------------------------------------------
__global__ void __launch_bounds__(256)
prep_kernel(const float* __restrict__ mean, const float* __restrict__ stddev,
            __bf16* __restrict__ Apack, float* __restrict__ cconst) {
  const int k = blockIdx.x;
  const int d = threadIdx.x;
  const size_t idx = (size_t)k * D_DIM + d;
  float m = mean[idx], s = stddev[idx];
  float iv  = 1.0f / (s * s);
  float miv = m * iv;
  __bf16* row = Apack + (size_t)k * 1024;   // 4 x 256 bf16
  __bf16 h = (__bf16)iv;
  row[d]       = h;  row[256 + d] = (__bf16)(iv - (float)h);
  h = (__bf16)miv;
  row[512 + d] = h;  row[768 + d] = (__bf16)(miv - (float)h);

  __shared__ float r1[256], r2[256];
  r1[d] = m * miv;
  r2[d] = __logf(s);
  __syncthreads();
  for (int off = 128; off > 0; off >>= 1) {
    if (d < off) { r1[d] += r1[d + off]; r2[d] += r2[d + off]; }
    __syncthreads();
  }
  if (d == 0) cconst[k] = -0.5f * r1[0] - r2[0];
}

// ---------------------------------------------------------------------------
// Kernel 2: async-LDS double-buffered bf16x3 WMMA + per-lane running top-32
//   grid = (B/64, KSB), block = 128.  4 waves share one cluster-tile stream
//   (staged via global_load_async_to_lds_b128); wave w owns rows w*16..w*16+15.
// ---------------------------------------------------------------------------
__global__ void __launch_bounds__(128)
gemm_topk_kernel(const float* __restrict__ x,
                 const __bf16* __restrict__ Apack,
                 const float* __restrict__ cconst,
                 uint2* __restrict__ cand) {
  __shared__ __align__(32) __bf16 s_xxh[ROWS_PER_BLK * D_DIM];
  __shared__ __align__(32) __bf16 s_xxl[ROWS_PER_BLK * D_DIM];
  __shared__ __align__(32) __bf16 s_mxh[ROWS_PER_BLK * D_DIM];
  __shared__ __align__(32) __bf16 s_mxl[ROWS_PER_BLK * D_DIM];
  __shared__ __align__(16) __bf16 s_A[2 * 16 * 1024];          // 2 x 32KB
  __shared__ uint2 s_top[WAVES_PER_BLK][32 * 33];

  const int tid   = threadIdx.x;
  const int btile = blockIdx.x;

  // stage x-side bf16 hi/lo of xx and -2x (64 rows, shared by the block)
  for (int i = tid; i < ROWS_PER_BLK * D_DIM; i += 128) {
    float xv = x[(size_t)(btile * ROWS_PER_BLK + (i >> 8)) * D_DIM + (i & 255)];
    float xx = xv * xv;
    float mx = -2.0f * xv;
    __bf16 h = (__bf16)xx; s_xxh[i] = h; s_xxl[i] = (__bf16)(xx - (float)h);
    h = (__bf16)mx;        s_mxh[i] = h; s_mxl[i] = (__bf16)(mx - (float)h);
  }

  const int wave = tid >> 5, lane = tid & 31;
  const int ln   = lane & 15, half = lane >> 4;
  const int cbase = blockIdx.y * (K_CL / KSB);

  uint2* slots = &s_top[wave][lane * 33];
  #pragma unroll
  for (int s = 0; s < TOPK; ++s) slots[s] = make_uint2(0xFF800000u, 0u);
  float rmin  = -__builtin_inff();
  int   rslot = 0;

  const unsigned ldsA = lds_byte_off(&s_A[0]);
  const unsigned voff = (unsigned)tid * 16u;
  const unsigned long long gA = (unsigned long long)(size_t)Apack;

  // prologue: prefetch stage 0 into buffer 0
  issue_async_stage(ldsA + voff, voff, gA + (unsigned long long)cbase * 2048ull);

  const int xrow = (wave * 16 + ln) * D_DIM + 8 * half;

  for (int st = 0; st < STAGES; ++st) {
    const int buf = st & 1;
    if (st + 1 < STAGES) {   // prefetch next stage into other buffer
      issue_async_stage(ldsA + (unsigned)(buf ^ 1) * 32768u + voff, voff,
                        gA + (unsigned long long)(cbase + (st + 1) * 16) * 2048ull);
      asm volatile("s_wait_asynccnt 0x10" ::: "memory");  // this stage's 16 done
    } else {
      asm volatile("s_wait_asynccnt 0x0" ::: "memory");
    }
    __syncthreads();   // all waves' chunks of this buffer landed

    v8f C = (v8f)(0.0f);
    const __bf16* ab = s_A + buf * 16384 + ln * 1024;   // this lane's cluster row
    #pragma unroll
    for (int dt = 0; dt < 8; ++dt) {
      const int d0 = dt * 32;
      v16bf Bxxh = loadA(s_xxh + xrow + d0);
      v16bf Bxxl = loadA(s_xxl + xrow + d0);
      v16bf Bmxh = loadA(s_mxh + xrow + d0);
      v16bf Bmxl = loadA(s_mxl + xrow + d0);
      const int da = d0 + 8 * half;
      v16bf Ah = loadA(ab + da);              // iv hi
      C = wmma_bf16(Ah, Bxxh, C);
      C = wmma_bf16(Ah, Bxxl, C);
      v16bf Al = loadA(ab + 256 + da);        // iv lo
      C = wmma_bf16(Al, Bxxh, C);
      v16bf Mh = loadA(ab + 512 + da);        // miv hi
      C = wmma_bf16(Mh, Bmxh, C);
      C = wmma_bf16(Mh, Bmxl, C);
      v16bf Ml = loadA(ab + 768 + da);        // miv lo
      C = wmma_bf16(Ml, Bmxh, C);
    }

    // epilogue: lp = -0.5*quad + cconst[k]; per-lane top-k update
    const int kb = cbase + st * 16 + 8 * half;
    const float4 c0 = *(const float4*)(cconst + kb);
    const float4 c1 = *(const float4*)(cconst + kb + 4);
    const float cc[8] = {c0.x, c0.y, c0.z, c0.w, c1.x, c1.y, c1.z, c1.w};
    #pragma unroll
    for (int r = 0; r < 8; ++r)
      topk_ins(slots, -0.5f * C[r] + cc[r], kb + r, rmin, rslot);

    __syncthreads();   // everyone done reading this buffer before reuse
  }

  const int brow = btile * ROWS_PER_BLK + wave * 16 + ln;
  const int list = blockIdx.y * 2 + half;
  uint2* dst = cand + ((size_t)brow * NLISTS + list) * TOPK;
  #pragma unroll
  for (int s = 0; s < TOPK; ++s) dst[s] = slots[s];
}

// ---------------------------------------------------------------------------
// Kernel 3: merge 64x32 candidates -> exact top-32, softmax, weighted gather
// ---------------------------------------------------------------------------
__global__ void __launch_bounds__(256)
merge_kernel(const uint2* __restrict__ cand, const float* __restrict__ outputs,
             float* __restrict__ out) {
  const int b = blockIdx.x;
  const int t = threadIdx.x;
  __shared__ float vals[NLISTS * TOPK];
  __shared__ int   idxs[NLISTS * TOPK];
  __shared__ float red[256];
  __shared__ int   wpos;
  __shared__ float selv[TOPK];
  __shared__ int   seli[TOPK];
  __shared__ float ew[TOPK];
  __shared__ float rsum;

  const uint2* src = cand + (size_t)b * (NLISTS * TOPK);
  for (int j = t; j < NLISTS * TOPK; j += 256) {
    uint2 p = src[j];
    vals[j] = __uint_as_float(p.x);
    idxs[j] = (int)p.y;
  }
  __syncthreads();

  const float NEGINF = -__builtin_inff();
  for (int it = 0; it < TOPK; ++it) {
    float lm = NEGINF;
    #pragma unroll
    for (int j = 0; j < 8; ++j) lm = fmaxf(lm, vals[t * 8 + j]);
    red[t] = lm;
    __syncthreads();
    for (int off = 128; off > 0; off >>= 1) {
      if (t < off) red[t] = fmaxf(red[t], red[t + off]);
      __syncthreads();
    }
    const float m = red[0];
    if (t == 0) wpos = 0x7fffffff;
    __syncthreads();
    #pragma unroll
    for (int j = 0; j < 8; ++j)
      if (vals[t * 8 + j] == m) atomicMin(&wpos, t * 8 + j);
    __syncthreads();
    if (t == 0) { selv[it] = m; seli[it] = idxs[wpos]; vals[wpos] = NEGINF; }
    __syncthreads();
  }

  if (t < TOPK) ew[t] = __expf(selv[t] - selv[0]);
  __syncthreads();
  if (t == 0) {
    float s = 0.f;
    for (int j = 0; j < TOPK; ++j) s += ew[j];
    rsum = 1.0f / s;
  }
  __syncthreads();

  float acc = 0.f;
  #pragma unroll 8
  for (int j = 0; j < TOPK; ++j)
    acc += ew[j] * outputs[(size_t)seli[j] * O_DIM + t];
  out[(size_t)b * O_DIM + t] = acc * rsum;
}

// ---------------------------------------------------------------------------
extern "C" void kernel_launch(void* const* d_in, const int* in_sizes, int n_in,
                              void* d_out, int out_size, void* d_ws, size_t ws_size,
                              hipStream_t stream) {
  (void)in_sizes; (void)n_in; (void)out_size; (void)ws_size;
  const float* x       = (const float*)d_in[0];
  const float* mean    = (const float*)d_in[1];
  const float* stddev  = (const float*)d_in[2];
  const float* outputs = (const float*)d_in[3];

  char* ws = (char*)d_ws;
  __bf16* Apack  = (__bf16*)(ws + OFF_A);
  float*  cconst = (float*)(ws + OFF_CC);
  uint2*  cand   = (uint2*)(ws + OFF_CAND);

  prep_kernel<<<dim3(K_CL), dim3(D_DIM), 0, stream>>>(mean, stddev, Apack, cconst);
  gemm_topk_kernel<<<dim3(B_ROWS / ROWS_PER_BLK, KSB), dim3(128), 0, stream>>>(
      x, Apack, cconst, cand);
  merge_kernel<<<dim3(B_ROWS), dim3(256), 0, stream>>>(cand, outputs,
                                                       (float*)d_out);
}
